// RIMCell_40888088658530
// MI455X (gfx1250) — matrix operand order repaired and checked by reading, not compiled, not run on hardware
//
#include <hip/hip_runtime.h>
#include <hip/hip_bf16.h>

// ---------------------------------------------------------------------------
// Problem constants (from the reference)
// ---------------------------------------------------------------------------
#define BB   2048
#define II   512
#define HH   512
#define UU   6
#define IKD  64
#define IVD  400
#define IVP  416          // IV padded to multiple of 32 (K padding)
#define NVP  448          // Wv output cols padded to multiple of 64
#define NCHD 4
#define CKD  32
#define CVD  512
#define KSEL 4

typedef __attribute__((ext_vector_type(16))) __bf16 v16bf;
typedef __attribute__((ext_vector_type(8)))  float  v8f;

union FragBF {
    v16bf v;
    unsigned int w[8];
};

__device__ __forceinline__ unsigned short f2bf(float f) {
    unsigned int x = __float_as_uint(f);
    x += 0x7FFFu + ((x >> 16) & 1u);     // round-to-nearest-even
    return (unsigned short)(x >> 16);
}

__device__ __forceinline__ float sigmoidf(float x) {
    return 1.0f / (1.0f + expf(-x));
}

// ---------------------------------------------------------------------------
// Weight prep: f32 (K x N) -> bf16 (Npad x Kout) transposed, zero padded.
// Cold path (once per launch), guards allowed here.
// ---------------------------------------------------------------------------
__global__ __launch_bounds__(256)
void transpose_cvt(const float* __restrict__ in, int ldin, long long sInZ,
                   int K, int N,
                   unsigned short* __restrict__ out, int ldout, long long sOutZ,
                   int Npad) {
    __shared__ float t[32][33];
    in  += (long long)blockIdx.z * sInZ;
    out += (long long)blockIdx.z * sOutZ;
    const int k0 = blockIdx.x * 32;
    const int n0 = blockIdx.y * 32;
#pragma unroll
    for (int i = 0; i < 4; ++i) {
        int r = threadIdx.y + i * 8, c = threadIdx.x;
        int k = k0 + r, n = n0 + c;
        t[r][c] = (k < K && n < N) ? in[(long long)k * ldin + n] : 0.f;
    }
    __syncthreads();
#pragma unroll
    for (int i = 0; i < 4; ++i) {
        int r = threadIdx.y + i * 8, c = threadIdx.x;
        int n = n0 + r, k = k0 + c;
        if (n < Npad && k < ldout)
            out[(long long)n * ldout + k] = f2bf(t[c][r]);
    }
}

// f32 (rows x ldin) -> bf16 (rows x ldout), zero fill cols >= ldin.
__global__ void cvt_pad(const float* __restrict__ in, int ldin, long long sInZ,
                        unsigned short* __restrict__ out, int ldout, long long sOutZ,
                        long long rows) {
    in  += (long long)blockIdx.z * sInZ;
    out += (long long)blockIdx.z * sOutZ;
    long long total = rows * ldout;
    for (long long idx = (long long)blockIdx.x * blockDim.x + threadIdx.x;
         idx < total; idx += (long long)gridDim.x * blockDim.x) {
        int k = (int)(idx % ldout);
        long long n = idx / ldout;
        float f = (k < ldin) ? in[n * ldin + k] : 0.f;
        out[idx] = f2bf(f);
    }
}

// ---------------------------------------------------------------------------
// bf16 WMMA GEMM: C[z] (+)= A[z](MxK bf16, row major) * Bt[z](NxK bf16)^T
//                 (+ bias1[z][n] + bias2[z][n], n < biasN)
// Requirements: M % 256 == 0, N % 64 == 0, K % 32 == 0 (guaranteed by caller).
//
// Block tile 256x64, K-step 32, 256 threads = 8 waves (4x2 grid of 64x32
// wave tiles) -> 8 v_wmma_f32_16x16x32_bf16 per wave per K-step.
// Staging uses gfx1250 GLOBAL_LOAD_ASYNC_TO_LDS_B128 (ASYNCcnt) into
// double-buffered LDS: the async DMA for tile i+1 overlaps the wmma chain
// for tile i; one s_wait_asynccnt + one barrier per K-step.
// ---------------------------------------------------------------------------
template <bool ACC>
__global__ __launch_bounds__(256)
void gemm_bf16(int N, int K,
               const unsigned short* __restrict__ A, int lda, long long sAz,
               const unsigned short* __restrict__ Bt, int ldb, long long sBz,
               float* __restrict__ C, int ldc, long long sCz,
               const float* __restrict__ bias1, long long sB1z,
               const float* __restrict__ bias2, long long sB2z, int biasN) {
    __shared__ __align__(16) unsigned short sA[2][256 * 40];
    __shared__ __align__(16) unsigned short sB[2][64 * 40];

    const int tid  = threadIdx.x;
    const int lane = tid & 31;
    const int wave = tid >> 5;
    const int wr   = wave >> 1;     // 0..3 -> 64-row slab
    const int wc   = wave & 1;      // 0..1 -> 32-col slab
    const int m0   = blockIdx.y * 256;
    const int n0   = blockIdx.x * 64;
    const int z    = blockIdx.z;

    A  += (long long)z * sAz;
    Bt += (long long)z * sBz;
    C  += (long long)z * sCz;

    v8f acc[4][2];
#pragma unroll
    for (int i = 0; i < 4; ++i)
#pragma unroll
        for (int j = 0; j < 2; ++j)
            acc[i][j] = (v8f){0.f,0.f,0.f,0.f,0.f,0.f,0.f,0.f};

    const int mrow = lane & 15;
    const int hi   = lane >> 4;
    const int sr   = tid >> 2;          // 0..63
    const int sc   = (tid & 3) * 8;     // 0,8,16,24 (bf16 cols; 16B chunks)

    // Async-DMA one 256x32 A tile + 64x32 B tile into LDS buffer `bufi`.
    auto stage = [&](int k0, int bufi) {
#pragma unroll
        for (int p = 0; p < 4; ++p) {
            int row = sr + p * 64;
            unsigned la = (unsigned)(size_t)&sA[bufi][row * 40 + sc];
            const unsigned short* g = &A[(long long)(m0 + row) * lda + k0 + sc];
            asm volatile("global_load_async_to_lds_b128 %0, %1, off"
                         :: "v"(la), "v"((unsigned long long)g) : "memory");
        }
        {
            unsigned lb = (unsigned)(size_t)&sB[bufi][sr * 40 + sc];
            const unsigned short* g = &Bt[(long long)(n0 + sr) * ldb + k0 + sc];
            asm volatile("global_load_async_to_lds_b128 %0, %1, off"
                         :: "v"(lb), "v"((unsigned long long)g) : "memory");
        }
    };

    const int nk = K >> 5;
    stage(0, 0);
    asm volatile("s_wait_asynccnt 0" ::: "memory");
    __syncthreads();

    for (int i = 0; i < nk; ++i) {
        const int cur = i & 1;
        if (i + 1 < nk) stage((i + 1) << 5, cur ^ 1);   // overlaps compute below

        const unsigned short* baseA = &sA[cur][(wr * 64 + mrow) * 40];
        const unsigned short* pb0   = &sB[cur][(wc * 32 + mrow) * 40];
        const unsigned short* pb1   = &sB[cur][(wc * 32 + 16 + mrow) * 40];
        FragBF b0, b1;
#pragma unroll
        for (int j = 0; j < 8; ++j) {
            int kb = hi * 16 + j * 2;
            b0.w[j] = *(const unsigned int*)&pb0[kb];
            b1.w[j] = *(const unsigned int*)&pb1[kb];
        }
#pragma unroll
        for (int ai = 0; ai < 4; ++ai) {
            FragBF a;
            const unsigned short* pa = baseA + ai * (16 * 40);
#pragma unroll
            for (int j = 0; j < 8; ++j) {
                int ka = hi * 16 + j * 2;   // A 16-bit layout (contiguous k, matches B)
                a.w[j] = *(const unsigned int*)&pa[ka];
            }
            acc[ai][0] = __builtin_amdgcn_wmma_f32_16x16x32_bf16(
                false, a.v, false, b0.v, (short)0, acc[ai][0], false, false);
            acc[ai][1] = __builtin_amdgcn_wmma_f32_16x16x32_bf16(
                false, a.v, false, b1.v, (short)0, acc[ai][1], false, false);
        }
        if (i + 1 < nk) asm volatile("s_wait_asynccnt 0" ::: "memory");
        __syncthreads();
    }

    // ---- write back (no guards: M,N multiples of tile) ----
    const int c0 = n0 + wc * 32 + mrow;
    const int c1 = c0 + 16;
    float bs0 = 0.f, bs1 = 0.f;
    if (bias1) {
        if (c0 < biasN) bs0 += bias1[(long long)z * sB1z + c0];
        if (c1 < biasN) bs1 += bias1[(long long)z * sB1z + c1];
    }
    if (bias2) {
        if (c0 < biasN) bs0 += bias2[(long long)z * sB2z + c0];
        if (c1 < biasN) bs1 += bias2[(long long)z * sB2z + c1];
    }
    const int rbase = m0 + wr * 64 + hi * 8;
#pragma unroll
    for (int ai = 0; ai < 4; ++ai) {
#pragma unroll
        for (int j = 0; j < 8; ++j) {
            int r = rbase + ai * 16 + j;
            long long o0 = (long long)r * ldc + c0;
            long long o1 = (long long)r * ldc + c1;
            float v0 = acc[ai][0][j] + bs0;
            float v1 = acc[ai][1][j] + bs1;
            if (ACC) { v0 += C[o0]; v1 += C[o1]; }
            C[o0] = v0;
            C[o1] = v1;
        }
    }
}

// ---------------------------------------------------------------------------
// Router: scores -> top-4 mask -> 2-token softmax -> mixed inputs (bf16, padded)
// ---------------------------------------------------------------------------
__global__ __launch_bounds__(64)
void router_kernel(const float* __restrict__ qin, const float* __restrict__ kcat,
                   const float* __restrict__ bk, const float* __restrict__ bv,
                   float* __restrict__ maskb, unsigned short* __restrict__ inputs_bf) {
    const int b = blockIdx.x;
    const int t = threadIdx.x;
    __shared__ float s_q[UU * IKD], s_k0[IKD], s_bk[IKD];
    __shared__ float s_sc[UU * 2], s_mask[UU], s_p[UU * 2];

#pragma unroll
    for (int u = 0; u < UU; ++u)
        s_q[u * IKD + t] = qin[(long long)b * (UU * IKD) + u * IKD + t];
    s_k0[t] = kcat[(long long)b * 512 + t];   // x@Wk + bk (token 0)
    s_bk[t] = bk[t];                          // token 1 key = bk (x2 row is zero)
    __syncthreads();

    if (t < UU * 2) {
        int u = t >> 1, tok = t & 1;
        const float* kk = tok ? s_bk : s_k0;
        float d = 0.f;
        for (int j = 0; j < IKD; ++j) d += s_q[u * IKD + j] * kk[j];
        s_sc[t] = d * 0.125f;                 // 1/sqrt(IK)
    }
    __syncthreads();

    if (t < UU) {
        float s0 = s_sc[t * 2];
        int cnt = 0;
        for (int v = 0; v < UU; ++v) {
            float sv = s_sc[v * 2];
            if (sv > s0 || (sv == s0 && v < t)) ++cnt;  // jax top_k tie order
        }
        float m = (cnt < KSEL) ? 1.f : 0.f;
        s_mask[t] = m;
        maskb[(long long)b * UU + t] = m;
        float a = s_sc[t * 2], c = s_sc[t * 2 + 1];
        float mx = fmaxf(a, c);
        float e0 = expf(a - mx), e1 = expf(c - mx);
        float inv = 1.f / (e0 + e1);
        s_p[t * 2] = e0 * inv;
        s_p[t * 2 + 1] = e1 * inv;
    }
    __syncthreads();

    for (int j = t; j < IVP; j += 64) {
        float v0 = (j < IVD) ? kcat[(long long)b * 512 + IKD + j] : 0.f;
        float v1 = (j < IVD) ? bv[j] : 0.f;
#pragma unroll
        for (int u = 0; u < UU; ++u)
            inputs_bf[((long long)b * UU + u) * IVP + j] =
                f2bf(s_mask[u] * (s_p[u * 2] * v0 + s_p[u * 2 + 1] * v1));
    }
}

// ---------------------------------------------------------------------------
// LSTM pointwise: gates (u,b,2048) -> c_new, h_rnn (f32) + h_rnn (bf16)
// ---------------------------------------------------------------------------
__global__ void lstm_kernel(const float* __restrict__ gates,
                            const float* __restrict__ cs,
                            float* __restrict__ cnew, float* __restrict__ hrnn,
                            unsigned short* __restrict__ hrnn_bf, int total) {
    int idx = blockIdx.x * blockDim.x + threadIdx.x;
    if (idx >= total) return;
    int d = idx & 511;
    int u = (idx >> 9) % UU;
    int b = idx / (UU * HH);
    long long gb = ((long long)u * BB + b) * 2048;
    float si = sigmoidf(gates[gb + d]);
    float sf = sigmoidf(gates[gb + 512 + d]);
    float gg = tanhf(gates[gb + 1024 + d]);
    float so = sigmoidf(gates[gb + 1536 + d]);
    float c = sf * cs[idx] + si * gg;
    float h = so * tanhf(c);
    cnew[idx] = c;
    hrnn[idx] = h;
    hrnn_bf[idx] = f2bf(h);
}

// ---------------------------------------------------------------------------
// Communication attention: 4 heads x (6x6) over units, masked softmax.
// One 128-thread block (4 waves, one head each) per batch element.
// Emits ctx directly in bf16 for the Wo_c GEMM.
// ---------------------------------------------------------------------------
__global__ __launch_bounds__(128)
void attn_kernel(const float* __restrict__ qc, const float* __restrict__ kc,
                 const float* __restrict__ vc, const float* __restrict__ maskb,
                 unsigned short* __restrict__ ctx_bf) {
    const int b = blockIdx.x;
    const int tid = threadIdx.x;
    const int h = tid >> 5;
    const int lane = tid & 31;
    __shared__ float s_att[NCHD][UU][UU];
    __shared__ float s_mask[UU];
    if (tid < UU) s_mask[tid] = maskb[(long long)b * UU + tid];
    __syncthreads();

    const float* qb = qc + (long long)b * (UU * NCHD * CKD);
    const float* kb = kc + (long long)b * (UU * NCHD * CKD);
    const float* vb = vc + (long long)b * (UU * NCHD * CVD);

#pragma unroll
    for (int u = 0; u < UU; ++u) {
#pragma unroll
        for (int n = 0; n < UU; ++n) {
            float p = qb[u * 128 + h * 32 + lane] * kb[n * 128 + h * 32 + lane];
            for (int off = 16; off > 0; off >>= 1) p += __shfl_down(p, off, 32);
            if (lane == 0) s_att[h][u][n] = p * 0.17677669529663687f; // 1/sqrt(32)
        }
    }
    __syncthreads();

    if (lane < UU) {
        float mx = -1e30f;
        for (int n = 0; n < UU; ++n) mx = fmaxf(mx, s_att[h][lane][n]);
        float e[UU], sum = 0.f;
        for (int n = 0; n < UU; ++n) { e[n] = expf(s_att[h][lane][n] - mx); sum += e[n]; }
        float inv = s_mask[lane] / sum;     // softmax * mask[u]
        for (int n = 0; n < UU; ++n) s_att[h][lane][n] = e[n] * inv;
    }
    __syncthreads();

    unsigned short* cb = ctx_bf + (long long)b * (UU * NCHD * CVD);
#pragma unroll
    for (int u = 0; u < UU; ++u) {
        for (int d0 = 0; d0 < CVD; d0 += 32) {
            int d = d0 + lane;
            float acc = 0.f;
#pragma unroll
            for (int n = 0; n < UU; ++n)
                acc += s_att[h][u][n] * vb[n * 2048 + h * CVD + d];
            cb[u * 2048 + h * CVD + d] = f2bf(acc);
        }
    }
}

// ---------------------------------------------------------------------------
// Final masked merge
// ---------------------------------------------------------------------------
__global__ void final_kernel(const float* __restrict__ P, const float* __restrict__ hrnn,
                             const float* __restrict__ cnew, const float* __restrict__ maskb,
                             const float* __restrict__ hs, const float* __restrict__ cs,
                             float* __restrict__ out, int total) {
    int idx = blockIdx.x * blockDim.x + threadIdx.x;
    if (idx >= total) return;
    int u = (idx >> 9) % UU;
    int b = idx / (UU * HH);
    float m = maskb[(long long)b * UU + u];
    out[idx] = m * (P[idx] + hrnn[idx]) + (1.f - m) * hs[idx];
    out[(long long)total + idx] = m * cnew[idx] + (1.f - m) * cs[idx];
}

// ---------------------------------------------------------------------------
// Launch
// ---------------------------------------------------------------------------
extern "C" void kernel_launch(void* const* d_in, const int* in_sizes, int n_in,
                              void* d_out, int out_size, void* d_ws, size_t ws_size,
                              hipStream_t stream) {
    (void)in_sizes; (void)n_in; (void)out_size; (void)ws_size;
    const float* x     = (const float*)d_in[0];
    const float* hs    = (const float*)d_in[1];
    const float* cs    = (const float*)d_in[2];
    const float* Wk    = (const float*)d_in[3];
    const float* bk    = (const float*)d_in[4];
    const float* Wv    = (const float*)d_in[5];
    const float* bv    = (const float*)d_in[6];
    const float* Wq_in = (const float*)d_in[7];
    const float* Wq_c  = (const float*)d_in[8];
    const float* Wk_c  = (const float*)d_in[9];
    const float* Wv_c  = (const float*)d_in[10];
    const float* Wo_c  = (const float*)d_in[11];
    const float* W_ih  = (const float*)d_in[12];
    const float* W_hh  = (const float*)d_in[13];
    const float* b_ih  = (const float*)d_in[14];
    const float* b_hh  = (const float*)d_in[15];

    // ---- workspace carve: f32 regions first, then bf16 regions ----
    float* fw = (float*)d_ws;
    float* kcat  = fw; fw += (long long)BB * 512;          // [Wk out | Wv out padded]
    float* qin   = fw; fw += (long long)BB * UU * IKD;
    float* maskb = fw; fw += (long long)BB * UU;
    float* gates = fw; fw += (long long)UU * BB * 2048;    // reused as vc later
    float* cnew  = fw; fw += (long long)BB * UU * HH;
    float* hrnn  = fw; fw += (long long)BB * UU * HH;
    float* qc    = fw; fw += (long long)BB * UU * 128;
    float* kc    = fw; fw += (long long)BB * UU * 128;
    float* Pbuf  = fw; fw += (long long)BB * UU * HH;
    float* vc    = gates;

    unsigned short* uw = (unsigned short*)fw;
    unsigned short* x_bf    = uw; uw += (long long)BB * II;
    unsigned short* hs_bf   = uw; uw += (long long)BB * UU * HH;
    unsigned short* WkT     = uw; uw += (long long)IKD * II;
    unsigned short* WvT     = uw; uw += (long long)NVP * II;
    unsigned short* WqinT   = uw; uw += (long long)UU * IKD * HH;
    unsigned short* WihB    = uw; uw += (long long)UU * 2048 * IVP;
    unsigned short* WhhB    = uw; uw += (long long)UU * 2048 * HH;
    unsigned short* WqcT    = uw; uw += (long long)UU * 128 * HH;
    unsigned short* WkcT    = uw; uw += (long long)UU * 128 * HH;
    unsigned short* WvcT    = uw; uw += (long long)UU * 2048 * HH;
    unsigned short* WocT    = uw; uw += (long long)UU * CVD * 2048;
    unsigned short* inpB    = uw; uw += (long long)BB * UU * IVP;
    unsigned short* hrnn_bf = uw; uw += (long long)BB * UU * HH;
    unsigned short* ctx_bf  = uw; uw += (long long)BB * UU * 2048;

    dim3 blk(256);
    dim3 tblk(32, 8);
    const int Mt = BB / 256;   // 8

    // ---- prep: activations + weights to bf16 (weights transposed to [N][K]) ----
    cvt_pad<<<dim3(1024, 1, 1), blk, 0, stream>>>(x, II, 0, x_bf, II, 0, BB);
    cvt_pad<<<dim3(4096, 1, 1), blk, 0, stream>>>(hs, HH, 0, hs_bf, HH, 0, (long long)BB * UU);
    transpose_cvt<<<dim3(16, 2, 1), tblk, 0, stream>>>(Wk, IKD, 0, II, IKD, WkT, II, 0, IKD);
    transpose_cvt<<<dim3(16, 14, 1), tblk, 0, stream>>>(Wv, IVD, 0, II, IVD, WvT, II, 0, NVP);
    transpose_cvt<<<dim3(16, 2, UU), tblk, 0, stream>>>(
        Wq_in, IKD, (long long)HH * IKD, HH, IKD, WqinT, HH, (long long)IKD * HH, IKD);
    cvt_pad<<<dim3(2048, 1, UU), blk, 0, stream>>>(
        W_ih, IVD, (long long)2048 * IVD, WihB, IVP, (long long)2048 * IVP, 2048);
    cvt_pad<<<dim3(2048, 1, UU), blk, 0, stream>>>(
        W_hh, HH, (long long)2048 * HH, WhhB, HH, (long long)2048 * HH, 2048);
    transpose_cvt<<<dim3(16, 4, UU), tblk, 0, stream>>>(
        Wq_c, 128, (long long)HH * 128, HH, 128, WqcT, HH, (long long)128 * HH, 128);
    transpose_cvt<<<dim3(16, 4, UU), tblk, 0, stream>>>(
        Wk_c, 128, (long long)HH * 128, HH, 128, WkcT, HH, (long long)128 * HH, 128);
    transpose_cvt<<<dim3(16, 64, UU), tblk, 0, stream>>>(
        Wv_c, 2048, (long long)HH * 2048, HH, 2048, WvcT, HH, (long long)2048 * HH, 2048);
    transpose_cvt<<<dim3(64, 16, UU), tblk, 0, stream>>>(
        Wo_c, CVD, (long long)2048 * CVD, 2048, CVD, WocT, 2048, (long long)CVD * 2048, CVD);

    // ---- 1) kcat = [x@Wk + bk | x@Wv + bv (padded)] ----
    gemm_bf16<false><<<dim3(1, Mt, 1), blk, 0, stream>>>(
        IKD, II, x_bf, II, 0, WkT, II, 0, kcat, 512, 0, bk, 0, nullptr, 0, IKD);
    gemm_bf16<false><<<dim3(NVP / 64, Mt, 1), blk, 0, stream>>>(
        NVP, II, x_bf, II, 0, WvT, II, 0, kcat + IKD, 512, 0, bv, 0, nullptr, 0, IVD);

    // ---- 2) q_lay[u] = hs_u @ Wq_in[u] ----
    gemm_bf16<false><<<dim3(1, Mt, UU), blk, 0, stream>>>(
        IKD, HH, hs_bf, UU * HH, HH, WqinT, HH, (long long)IKD * HH,
        qin, UU * IKD, IKD, nullptr, 0, nullptr, 0, 0);

    // ---- 3) router ----
    router_kernel<<<dim3(BB), dim3(64), 0, stream>>>(qin, kcat, bk, bv, maskb, inpB);

    // ---- 4) gates[u] = inputs_u @ W_ih[u]^T + hs_u @ W_hh[u]^T + b_ih + b_hh ----
    gemm_bf16<false><<<dim3(32, Mt, UU), blk, 0, stream>>>(
        2048, IVP, inpB, UU * IVP, IVP, WihB, IVP, (long long)2048 * IVP,
        gates, 2048, (long long)BB * 2048, b_ih, 2048, b_hh, 2048, 2048);
    gemm_bf16<true><<<dim3(32, Mt, UU), blk, 0, stream>>>(
        2048, HH, hs_bf, UU * HH, HH, WhhB, HH, (long long)2048 * HH,
        gates, 2048, (long long)BB * 2048, nullptr, 0, nullptr, 0, 0);

    // ---- 5) LSTM pointwise ----
    const int tot = BB * UU * HH;
    lstm_kernel<<<dim3((tot + 255) / 256), blk, 0, stream>>>(gates, cs, cnew, hrnn, hrnn_bf, tot);

    // ---- 6) communication q/k/v projections (vc reuses gates region) ----
    gemm_bf16<false><<<dim3(2, Mt, UU), blk, 0, stream>>>(
        128, HH, hrnn_bf, UU * HH, HH, WqcT, HH, (long long)128 * HH,
        qc, UU * 128, 128, nullptr, 0, nullptr, 0, 0);
    gemm_bf16<false><<<dim3(2, Mt, UU), blk, 0, stream>>>(
        128, HH, hrnn_bf, UU * HH, HH, WkcT, HH, (long long)128 * HH,
        kc, UU * 128, 128, nullptr, 0, nullptr, 0, 0);
    gemm_bf16<false><<<dim3(32, Mt, UU), blk, 0, stream>>>(
        2048, HH, hrnn_bf, UU * HH, HH, WvcT, HH, (long long)2048 * HH,
        vc, UU * 2048, 2048, nullptr, 0, nullptr, 0, 0);

    // ---- 7) masked multi-head attention over units (emits bf16 ctx) ----
    attn_kernel<<<dim3(BB), dim3(128), 0, stream>>>(qc, kc, vc, maskb, ctx_bf);

    // ---- 8) P[u] = ctx_u @ Wo_c[u] ----
    gemm_bf16<false><<<dim3(8, Mt, UU), blk, 0, stream>>>(
        CVD, 2048, ctx_bf, UU * 2048, 2048, WocT, 2048, (long long)CVD * 2048,
        Pbuf, UU * CVD, CVD, nullptr, 0, nullptr, 0, 0);

    // ---- 9) masked merge into (hs_out, cs_out) ----
    final_kernel<<<dim3((tot + 255) / 256), blk, 0, stream>>>(
        Pbuf, hrnn, cnew, maskb, hs, cs, (float*)d_out, tot);
}
